// DGConv2_77429670412663
// MI455X (gfx1250) — matrix operand order repaired and checked by reading, not compiled, not run on hardware
//
#include <hip/hip_runtime.h>
#include <hip/hip_bf16.h>

// Problem constants (compile-time, from the reference)
constexpr int kB  = 4;
constexpr int kC  = 64;     // channels
constexpr int kN  = 8192;   // points
constexpr int kO  = 128;    // output channels
constexpr int kK  = 20;     // spiral size (top-k)
constexpr int kD  = 2 * kC; // 128 = feature dim

typedef __attribute__((ext_vector_type(8)))  _Float16 v8h;
typedef __attribute__((ext_vector_type(16))) _Float16 v16h;
typedef __attribute__((ext_vector_type(8)))  float    v8f;

union V16U { v16h v; v8h h[2]; };

// ---------------------------------------------------------------------------
// gfx1250 async global->LDS copies (ASYNCcnt), with synchronous fallback.
// Probe-verified: arg0 is (AS1) v4i*, arg1 is the LDS destination.
// ---------------------------------------------------------------------------
#if defined(__has_builtin)
#if __has_builtin(__builtin_amdgcn_global_load_async_to_lds_b128) && \
    __has_builtin(__builtin_amdgcn_s_wait_asynccnt)
#define HAS_ASYNC_LDS 1
#endif
#endif
#ifndef HAS_ASYNC_LDS
#define HAS_ASYNC_LDS 0
#endif

typedef int v4i __attribute__((vector_size(16)));
typedef __attribute__((address_space(1))) v4i global_v4i;
typedef __attribute__((address_space(3))) v4i lds_v4i;

__device__ __forceinline__ void async_copy16(void* lds_dst, const void* gsrc) {
#if HAS_ASYNC_LDS
    __builtin_amdgcn_global_load_async_to_lds_b128((global_v4i*)gsrc, (lds_v4i*)lds_dst, 0, 0);
#else
    *(v8h*)lds_dst = *(const v8h*)gsrc;
#endif
}
__device__ __forceinline__ void async_wait5() {
#if HAS_ASYNC_LDS
    __builtin_amdgcn_s_wait_asynccnt(5);
#endif
}
__device__ __forceinline__ void async_wait0() {
#if HAS_ASYNC_LDS
    __builtin_amdgcn_s_wait_asynccnt(0);
#endif
}

// A-matrix fragment, 16x32 f16 (ISA 7.12.2): lane M = L%16,
// koff = (L<16 ? 0 : 8); elems 0..7 = K(koff..koff+7), 8..15 = K(koff+16..koff+23).
__device__ __forceinline__ v16h load_a_frag(const _Float16* row, int kchunk, int lane) {
    int koff = (lane < 16) ? 0 : 8;
    const _Float16* p = row + kchunk + koff;
    V16U u;
    u.h[0] = *(const v8h*)(p);
    u.h[1] = *(const v8h*)(p + 16);
    return u.v;
}

// B-matrix fragment, 32x16 f16: lane N = L%16, koff = (L<16 ? 0 : 8);
// elems 0..7 = K(koff..koff+7), 8..15 = K(koff+16..koff+23) (matches A).
__device__ __forceinline__ v16h load_b_frag(const _Float16* row, int kchunk, int lane) {
    int kbase = (lane < 16) ? 0 : 8;
    const _Float16* p = row + kchunk + kbase;
    V16U u;
    u.h[0] = *(const v8h*)(p);
    u.h[1] = *(const v8h*)(p + 16);
    return u.v;
}

// ---------------------------------------------------------------------------
// Kernel 0: prep. Point-major f16 copy xh[b][n][c] (contiguous 128B per point),
// f32 squared norms xn[b][n], f16 copies of W1/W2.
// ---------------------------------------------------------------------------
__global__ __launch_bounds__(256) void prep_kernel(const float* __restrict__ x,
                                                   const float* __restrict__ W1,
                                                   const float* __restrict__ W2,
                                                   _Float16* __restrict__ xh,
                                                   float* __restrict__ xn,
                                                   _Float16* __restrict__ w1h,
                                                   _Float16* __restrict__ w2h) {
    int g = blockIdx.x * 256 + threadIdx.x;   // point id over B*N
    int b = g / kN, n = g % kN;
    const float* xb = x + (size_t)b * kC * kN;
    float s = 0.f;
    _Float16 tmp[kC];
#pragma unroll
    for (int c = 0; c < kC; ++c) {
        float v = xb[(size_t)c * kN + n];
        s += v * v;
        tmp[c] = (_Float16)v;
    }
    v8h* dst = (v8h*)(xh + (size_t)g * kC);
#pragma unroll
    for (int i = 0; i < 8; ++i) dst[i] = ((const v8h*)tmp)[i];
    xn[g] = s;
    if (blockIdx.x == 0) {
        for (int e = threadIdx.x; e < kO * kD; e += 256) {
            w1h[e] = (_Float16)W1[e];
            w2h[e] = (_Float16)W2[e];
        }
    }
}

// ---------------------------------------------------------------------------
// Kernel 1: KNN top-20 via WMMA gram tiles, async double-buffered staging.
// Block = (batch b, 128-query tile). 256 threads = 8 waves.
// ---------------------------------------------------------------------------
constexpr int kQT = 128;  // queries per block
constexpr int kCT = 128;  // candidates per LDS tile
constexpr int kDP = 132;  // dist pitch (conflict-free column reads)
constexpr int kTILES = kN / kCT;

struct MergeBuf { float kd[kQT][kK]; int ki[kQT][kK]; };
union  ScanU    { float dist[kCT][kDP]; MergeBuf m; };

// Stage one 128-point tile: 4x b128 per thread for the f16 rows (16KB) plus
// 1x b128 per thread for the norms (redundant across waves so every wave
// issues exactly 5 async instructions -> ASYNCcnt bookkeeping is uniform).
__device__ __forceinline__ void stage_tile(_Float16 (*dstC)[kC], float* dstN,
                                           const _Float16* xhb, const float* xnb,
                                           int m0, int t) {
    const char* src = (const char*)(xhb + (size_t)m0 * kC);
#pragma unroll
    for (int i = 0; i < 4; ++i) {
        int chunk = t + i * 256;  // 0..1023, 16B each
        async_copy16((char*)&dstC[0][0] + chunk * 16, src + chunk * 16);
    }
    int nc = t & 31;              // 0..31, 16B each (512B of norms)
    async_copy16((char*)dstN + nc * 16, (const char*)(xnb + m0) + nc * 16);
}

__global__ __launch_bounds__(256) void knn_topk_kernel(const _Float16* __restrict__ xh,
                                                       const float* __restrict__ xn,
                                                       int* __restrict__ knn_idx) {
    __shared__ __align__(16) _Float16 qs[kQT][kC];       // 16 KB queries  [q][c]
    __shared__ __align__(16) _Float16 cs[2][kCT][kC];    // 32 KB candidates, double buffered
    __shared__ __align__(16) float    qn[kQT];
    __shared__ __align__(16) float    cn[2][kCT];
    __shared__ __align__(16) ScanU    su;                // 66 KB dist / merge lists

    const int b  = blockIdx.x / (kN / kQT);
    const int q0 = (blockIdx.x % (kN / kQT)) * kQT;
    const _Float16* xhb = xh + (size_t)b * kN * kC;
    const float*    xnb = xn + (size_t)b * kN;

    const int t    = threadIdx.x;
    const int lane = t & 31;
    const int wave = t >> 5;

    // Stage queries + tile 0 (async, in-flight while we fall into the loop).
    stage_tile(qs, qn, xhb, xnb, q0, t);
    stage_tile(cs[0], cn[0], xhb, xnb, 0, t);

    // Per-thread top-k: thread t owns query (t&127), candidate half (t>>7).
    const int q    = t & 127;
    const int jb   = (t >> 7) * 64;  // scan range [jb, jb+64)
    float kd[kK];
    int   ki[kK];
#pragma unroll
    for (int i = 0; i < kK; ++i) { kd[i] = 3.4e38f; ki[i] = 0; }
    float worst = 3.4e38f;

    for (int it = 0; it < kTILES; ++it) {
        const int cur = it & 1;
        if (it + 1 < kTILES) {
            stage_tile(cs[cur ^ 1], cn[cur ^ 1], xhb, xnb, (it + 1) * kCT, t);
            async_wait5();              // tile `it` landed; tile it+1 in flight
        } else {
            async_wait0();
        }
        __syncthreads();

        // Gram tiles: wave w owns query rows [w*16, w*16+16), all 8 cand tiles.
        const int qrow = wave * 16;
        for (int mt = 0; mt < kCT / 16; ++mt) {
            v8f acc = {};
#pragma unroll
            for (int kc = 0; kc < kC; kc += 32) {
                v16h a  = load_a_frag(&qs[qrow + (lane & 15)][0], kc, lane);
                v16h bf = load_b_frag(&cs[cur][mt * 16 + (lane & 15)][0], kc, lane);
                acc = __builtin_amdgcn_wmma_f32_16x16x32_f16(
                        false, a, false, bf, (short)0, acc, false, false);
            }
            const int nloc = mt * 16 + (lane & 15);
            const float cnv = cn[cur][nloc];
#pragma unroll
            for (int r = 0; r < 8; ++r) {
                int qm = qrow + ((lane < 16) ? r : r + 8);
                su.dist[nloc][qm] = qn[qm] + cnv - 2.0f * acc[r];
            }
        }
        __syncthreads();

        // Scan: all 256 threads, 64 candidates each.
        for (int j = jb; j < jb + 64; ++j) {
            float d = su.dist[j][q];
            if (d < worst) {
                int slot = 0; float w2 = -3.4e38f;
#pragma unroll
                for (int i = 0; i < kK; ++i)
                    if (kd[i] > w2) { w2 = kd[i]; slot = i; }
#pragma unroll
                for (int i = 0; i < kK; ++i)
                    if (i == slot) { kd[i] = d; ki[i] = it * kCT + j; }
                worst = -3.4e38f;
#pragma unroll
                for (int i = 0; i < kK; ++i) worst = fmaxf(worst, kd[i]);
            }
        }
    }

    // Merge the two candidate-half lists per query (aliased over dist).
    __syncthreads();
    if (t >= 128) {
#pragma unroll
        for (int i = 0; i < kK; ++i) { su.m.kd[q][i] = kd[i]; su.m.ki[q][i] = ki[i]; }
    }
    __syncthreads();
    if (t < 128) {
#pragma unroll
        for (int i = 0; i < kK; ++i) {
            float d = su.m.kd[q][i];
            int  id = su.m.ki[q][i];
            if (d < worst) {
                int slot = 0; float w2 = -3.4e38f;
#pragma unroll
                for (int i2 = 0; i2 < kK; ++i2)
                    if (kd[i2] > w2) { w2 = kd[i2]; slot = i2; }
#pragma unroll
                for (int i2 = 0; i2 < kK; ++i2)
                    if (i2 == slot) { kd[i2] = d; ki[i2] = id; }
                worst = -3.4e38f;
#pragma unroll
                for (int i2 = 0; i2 < kK; ++i2) worst = fmaxf(worst, kd[i2]);
            }
        }
        int* o = knn_idx + ((size_t)b * kN + q0 + q) * kK;
#pragma unroll
        for (int i = 0; i < kK; ++i) o[i] = ki[i];
    }
}

// ---------------------------------------------------------------------------
// Kernel 2: feats -> W1 GEMM -> softmax(k) -> gated reduce -> W2 GEMM.
// Block = (batch b, 4 points) => 80 feat rows = 5 WMMA M-tiles. 256 threads.
// All global traffic is async row copies from the point-major f16 mirror.
// ---------------------------------------------------------------------------
constexpr int kNT   = 4;          // points per block
constexpr int kROWS = kNT * kK;   // 80

__global__ __launch_bounds__(256) void dgconv_kernel(const _Float16* __restrict__ xh,
                                                     const _Float16* __restrict__ w1h,
                                                     const _Float16* __restrict__ w2h,
                                                     const int* __restrict__ knn_idx,
                                                     float* __restrict__ out) {
    __shared__ __align__(16) _Float16 sW[kO][kD];           // 32 KB: W1 then W2
    __shared__ __align__(16) _Float16 nbuf[kROWS + kNT][kC];// 10.5 KB: 80 nbr rows + 4 centers
    __shared__ __align__(16) _Float16 sF[kROWS][kD];        // 20 KB feats f16
    __shared__ __align__(16) float    sH[kROWS][kD];        // 40 KB h (f32)
    __shared__ __align__(16) _Float16 sG[16][kD];           //  4 KB gated sums (padded)
    __shared__ int sidx[kROWS];

    const int b  = blockIdx.x / (kN / kNT);
    const int n0 = (blockIdx.x % (kN / kNT)) * kNT;
    const _Float16* xhb = xh + (size_t)b * kN * kC;

    const int t    = threadIdx.x;
    const int lane = t & 31;
    const int wave = t >> 5;

    if (t < kROWS)
        sidx[t] = knn_idx[((size_t)b * kN + n0 + t / kK) * kK + (t % kK)];
    for (int e = t; e < 16 * kD; e += 256) sG[0][e] = (_Float16)0.0f;
    __syncthreads();   // sidx visible before gather issues

    // Async stage: W1 (8 x b128 per thread) + 84 rows of 128B (nbrs + centers).
#pragma unroll
    for (int i = 0; i < 8; ++i) {
        int chunk = t + i * 256;  // < 2048
        async_copy16((char*)&sW[0][0] + chunk * 16, (const char*)w1h + chunk * 16);
    }
    for (int task = t; task < (kROWS + kNT) * 8; task += 256) {
        int row = task >> 3, ch = task & 7;
        int nsrc = (row < kROWS) ? sidx[row] : (n0 + row - kROWS);
        const char* src = (const char*)(xhb + (size_t)nsrc * kC) + ch * 16;
        async_copy16((char*)&nbuf[row][0] + ch * 16, src);
    }
    async_wait0();
    __syncthreads();

    // Build feats = [nbr - center, center] (f16, from LDS rows).
    for (int e = t; e < kROWS * kD; e += 256) {
        int row = e / kD, c = e % kD;
        int p = row / kK;
        _Float16 v;
        if (c < kC) v = (_Float16)((float)nbuf[row][c] - (float)nbuf[kROWS + p][c]);
        else        v = nbuf[kROWS + p][c - kC];
        sF[row][c] = v;
    }
    __syncthreads();

    // Phase A: h = feats * W1^T. Wave w owns d-tile w (8 tiles), 5 M-tiles each.
    {
        const int d0 = wave * 16;
        for (int mt = 0; mt < kROWS / 16; ++mt) {
            v8f acc = {};
#pragma unroll
            for (int kc = 0; kc < kD; kc += 32) {
                v16h a  = load_a_frag(&sF[mt * 16 + (lane & 15)][0], kc, lane);
                v16h bf = load_b_frag(&sW[d0 + (lane & 15)][0], kc, lane);
                acc = __builtin_amdgcn_wmma_f32_16x16x32_f16(
                        false, a, false, bf, (short)0, acc, false, false);
            }
            const int dcol = d0 + (lane & 15);
#pragma unroll
            for (int r = 0; r < 8; ++r) {
                int row = mt * 16 + ((lane < 16) ? r : r + 8);
                sH[row][dcol] = acc[r];
            }
        }
    }
    __syncthreads();

    // Phase B: async-reload W2 into sW while doing softmax(k) + gated reduce.
#pragma unroll
    for (int i = 0; i < 8; ++i) {
        int chunk = t + i * 256;
        async_copy16((char*)&sW[0][0] + chunk * 16, (const char*)w2h + chunk * 16);
    }
    for (int col = t; col < kNT * kD; col += 256) {
        int p = col / kD, d = col % kD;
        float mx = -3.4e38f;
#pragma unroll
        for (int k = 0; k < kK; ++k) mx = fmaxf(mx, sH[p * kK + k][d]);
        float ex[kK];
        float ssum = 0.f;
#pragma unroll
        for (int k = 0; k < kK; ++k) {
            float e2 = __expf(sH[p * kK + k][d] - mx);
            ex[k] = e2; ssum += e2;
        }
        float inv = 1.0f / ssum;
        float g = 0.f;
#pragma unroll
        for (int k = 0; k < kK; ++k) g += (float)sF[p * kK + k][d] * ex[k];
        sG[p][d] = (_Float16)(g * inv);
    }
    async_wait0();
    __syncthreads();

    // Phase C: out = W2 * g. Wave w owns o-tile w; 16 cols = 4 real points + pad.
    {
        const int o0 = wave * 16;
        v8f acc = {};
#pragma unroll
        for (int kc = 0; kc < kD; kc += 32) {
            v16h a  = load_a_frag(&sW[o0 + (lane & 15)][0], kc, lane);
            v16h bf = load_b_frag(&sG[lane & 15][0], kc, lane);
            acc = __builtin_amdgcn_wmma_f32_16x16x32_f16(
                    false, a, false, bf, (short)0, acc, false, false);
        }
        const int p = lane & 15;
        if (p < kNT) {
#pragma unroll
            for (int r = 0; r < 8; ++r) {
                int o = o0 + ((lane < 16) ? r : r + 8);
                out[(size_t)b * kO * kN + (size_t)o * kN + n0 + p] = acc[r];
            }
        }
    }
}

// ---------------------------------------------------------------------------
extern "C" void kernel_launch(void* const* d_in, const int* in_sizes, int n_in,
                              void* d_out, int out_size, void* d_ws, size_t ws_size,
                              hipStream_t stream) {
    const float* x  = (const float*)d_in[0];
    const float* W1 = (const float*)d_in[1];
    const float* W2 = (const float*)d_in[2];
    // d_in[3] = spiral_size (compile-time kK = 20)

    char* ws = (char*)d_ws;
    _Float16* xh  = (_Float16*)(ws);                 // 4,194,304 B
    float*    xn  = (float*)   (ws + 4194304);       //   131,072 B
    _Float16* w1h = (_Float16*)(ws + 4325376);       //    32,768 B
    _Float16* w2h = (_Float16*)(ws + 4358144);       //    32,768 B
    int*      knn = (int*)     (ws + 4390912);       // 2,621,440 B   (total ~6.7 MB)

    prep_kernel<<<kB * kN / 256, 256, 0, stream>>>(x, W1, W2, xh, xn, w1h, w2h);
    knn_topk_kernel<<<kB * (kN / kQT), 256, 0, stream>>>(xh, xn, knn);
    dgconv_kernel<<<kB * (kN / kNT), 256, 0, stream>>>(xh, w1h, w2h, knn, (float*)d_out);
}
